// ColorLayer_45329084842180
// MI455X (gfx1250) — matrix operand order repaired and checked by reading, not compiled, not run on hardware
//
#include <hip/hip_runtime.h>

// ---------------------------------------------------------------------------
// Modulated 3x3 conv (StyleGAN2 style), B=8 H=W=128 C=F=256.
// y[b,h,w,f] = rdemod[b,f] * sum_{ky,kx,c} K[ky,kx,c,f] * xs[b,h+ky-1,w+kx-1,c]
//   xs = x * (style+1)  (bf16),  rdemod = 1/sqrt(sum (style+1)^2 * K^2 + 1e-8)
// Compute-bound (~600 flop/byte) -> bf16 WMMA path (v_wmma_f32_16x16x32_bf16).
// Conv kernel: WG = 64 px x 256 filters, 8 waves, each wave 4x2 register tile
// (1.5 loads per WMMA; weight traffic amortized over 64 pixels).
// LDS staging via global_load_async_to_lds_b128 (ASYNCcnt-tracked).
// ---------------------------------------------------------------------------

typedef __attribute__((ext_vector_type(16))) __bf16 v16bf;
typedef __attribute__((ext_vector_type(8)))  float  v8f;
typedef __attribute__((__vector_size__(4 * sizeof(int)))) int vi4;
typedef __attribute__((address_space(1))) vi4* gvi4p;   // global int4*
typedef __attribute__((address_space(3))) vi4* lvi4p;   // LDS int4*

#define Bn   8
#define Hd   128
#define Wd   128
#define Cc   256
#define Ff   256
#define MT   64           // output pixels per workgroup (along W)
#define SW   66           // staged strip width (MT + 2 halo)
#define PIXS 264          // ushorts per pixel row in LDS (256 + 8 pad = 528B)

#if __has_builtin(__builtin_amdgcn_global_load_async_to_lds_b128) && \
    __has_builtin(__builtin_amdgcn_s_wait_asynccnt)
#define USE_ASYNC_LDS 1
#else
#define USE_ASYNC_LDS 0
#endif

__device__ __forceinline__ unsigned short f2bf(float f) {
    unsigned u = __builtin_bit_cast(unsigned, f);
    unsigned r = u + 0x7FFFu + ((u >> 16) & 1u);   // round-to-nearest-even
    return (unsigned short)(r >> 16);
}

struct U32x8 { uint4 lo, hi; };

__device__ __forceinline__ v16bf ld_frag_lds(const unsigned short* p, int step) {
    U32x8 t;
    t.lo = *(const uint4*)(p);
    t.hi = *(const uint4*)(p + step);
    return __builtin_bit_cast(v16bf, t);
}

// --- kernel 1: xs = bf16( x * (style + 1) ), 8 elems/thread -----------------
__global__ void __launch_bounds__(256) k_prescale(const float* __restrict__ x,
                                                  const float* __restrict__ style,
                                                  unsigned short* __restrict__ xs,
                                                  int total8) {
    int t = blockIdx.x * 256 + threadIdx.x;
    if (t >= total8) return;
    long i0 = (long)t << 3;
    int c = (int)(i0 & (Cc - 1));
    int b = (int)(i0 >> 22);                    // H*W*C = 2^22
    const float4* xp = (const float4*)(x + i0);
    float4 v0 = xp[0], v1 = xp[1];
    const float* sp = style + b * Cc + c;
    float r[8];
    r[0] = v0.x * (sp[0] + 1.f); r[1] = v0.y * (sp[1] + 1.f);
    r[2] = v0.z * (sp[2] + 1.f); r[3] = v0.w * (sp[3] + 1.f);
    r[4] = v1.x * (sp[4] + 1.f); r[5] = v1.y * (sp[5] + 1.f);
    r[6] = v1.z * (sp[6] + 1.f); r[7] = v1.w * (sp[7] + 1.f);
    uint4 o;
    o.x = (unsigned)f2bf(r[0]) | ((unsigned)f2bf(r[1]) << 16);
    o.y = (unsigned)f2bf(r[2]) | ((unsigned)f2bf(r[3]) << 16);
    o.z = (unsigned)f2bf(r[4]) | ((unsigned)f2bf(r[5]) << 16);
    o.w = (unsigned)f2bf(r[6]) | ((unsigned)f2bf(r[7]) << 16);
    *(uint4*)(xs + i0) = o;
}

// --- kernel 2: pack weights into per-lane B-fragment layout -----------------
// wpk[t][q][f][h][k] = bf16( K[t, q*32 + (k>>3)*16 + h*8 + (k&7), f] ), t=tap, q=chan chunk
__global__ void __launch_bounds__(256) k_wpack(const float* __restrict__ kern,
                                               unsigned short* __restrict__ wpk,
                                               int total) {
    int g = blockIdx.x * 256 + threadIdx.x;
    if (g >= total) return;
    int kk = g & 15;
    int h  = (g >> 4) & 1;
    int f  = (g >> 5) & (Ff - 1);
    int q  = (g >> 13) & 7;
    int t  = g >> 16;                           // 0..8
    int c  = q * 32 + ((kk >> 3) << 4) + h * 8 + (kk & 7);
    wpk[g] = f2bf(kern[(((long)t * Cc + c) << 8) + f]);
}

// --- kernel 3: rdemod[b,f] = 1/sqrt( sum_c (style+1)^2 * sum_t K^2 + 1e-8 ) -
__global__ void __launch_bounds__(256) k_demod(const float* __restrict__ kern,
                                               const float* __restrict__ style,
                                               float* __restrict__ rd) {
    int g = blockIdx.x * 256 + threadIdx.x;     // B*F = 2048
    int f = g & (Ff - 1);
    int b = g >> 8;
    float acc = 0.f;
    for (int c = 0; c < Cc; ++c) {
        float s = style[b * Cc + c] + 1.f;
        float ks = 0.f;
        #pragma unroll
        for (int t = 0; t < 9; ++t) {
            float w = kern[(((long)t * Cc + c) << 8) + f];
            ks += w * w;
        }
        acc += s * s * ks;
    }
    rd[g] = 1.0f / __builtin_sqrtf(acc + 1e-8f);
}

// --- kernel 4: implicit-GEMM conv with WMMA, 4x2 wave tiling ----------------
// Grid: b(8) x h(128) x wtile(2) = 2048 WGs.  WG covers 64 px x 256 filters.
// Wave wv owns filters [wv*32, wv*32+32) over all 64 px: 4 M-tiles x 2 N-tiles.
__global__ void __launch_bounds__(256) k_conv(const unsigned short* __restrict__ xs,
                                              const unsigned short* __restrict__ wpk,
                                              const float* __restrict__ rdemod,
                                              float* __restrict__ out) {
    __shared__ unsigned short ls[3 * SW * PIXS];   // 104,544 B

    const int tid  = threadIdx.x;
    const int gid  = blockIdx.x;
    const int wblk = gid & 1;                 // 2 pixel blocks of 64
    const int h    = (gid >> 1) & (Hd - 1);
    const int b    = gid >> 8;
    const int w0   = wblk * MT;

    // Stage 3 rows x 66 px x 256 ch of xs (zero-padded halo) into LDS.
    for (int j = tid; j < 3 * SW * 32; j += 256) {
        int r   = j / (SW * 32);
        int rem = j - r * (SW * 32);
        int p   = rem >> 5;
        int cc  = rem & 31;
        int hs  = h - 1 + r;
        int ws  = w0 - 1 + p;
        unsigned short* dst = &ls[(r * SW + p) * PIXS + (cc << 3)];
        if ((unsigned)hs < (unsigned)Hd && (unsigned)ws < (unsigned)Wd) {
            const unsigned short* src =
                xs + ((((long)b * Hd + hs) * Wd + ws) << 8) + (cc << 3);
#if USE_ASYNC_LDS
            __builtin_amdgcn_global_load_async_to_lds_b128(
                (gvi4p)(void*)src, (lvi4p)(void*)dst, 0, 0);
#else
            *(uint4*)dst = *(const uint4*)src;
#endif
        } else {
            *(uint4*)dst = make_uint4(0u, 0u, 0u, 0u);
        }
    }
#if USE_ASYNC_LDS
    __builtin_amdgcn_s_wait_asynccnt(0);
#endif
    __syncthreads();

    const int lane   = tid & 31;
    const int wv     = tid >> 5;               // 8 waves -> filter groups of 32
    const int fbase  = wv * 32;
    const int mrow   = lane & 15;              // A: row M
    const int hilane = lane >> 4;              // 0: K 0-7,16-23, 1: K 8-15,24-31
    const int koffA  = hilane << 3;            // A lane chan offset (0 or 8)
    const int nf     = lane & 15;              // B/D: column N

    v8f acc[4][2];
    #pragma unroll
    for (int m = 0; m < 4; ++m) {
        acc[m][0] = (v8f){0.f,0.f,0.f,0.f,0.f,0.f,0.f,0.f};
        acc[m][1] = (v8f){0.f,0.f,0.f,0.f,0.f,0.f,0.f,0.f};
    }

    // Per-lane B base: [t][q][f][h][16] -> ((t*8+q)*256 + f)*32 + h*16 ushorts
    const unsigned short* bb0 = wpk + (((long)(fbase + nf)) << 5) + (hilane << 4);
    const unsigned short* bb1 = bb0 + (16 << 5);            // +16 filters

    #pragma unroll 1
    for (int t = 0; t < 9; ++t) {
        const int ky = t / 3;
        const int kx = t - ky * 3;
        const int prow = (ky * SW + mrow + kx) * PIXS;      // M-tile 0 row
        const unsigned short* bt0 = bb0 + ((long)t << 16);  // 8*256*32/tap
        const unsigned short* bt1 = bb1 + ((long)t << 16);
        #pragma unroll
        for (int q = 0; q < 8; ++q) {
            const int aoff = prow + (q << 5) + koffA;
            v16bf A[4];
            #pragma unroll
            for (int m = 0; m < 4; ++m)                     // M-tiles: +16 px
                A[m] = ld_frag_lds(&ls[aoff + m * (16 * PIXS)], 16);
            v16bf B0 = ld_frag_lds(bt0 + ((long)q << 13), 8);
            v16bf B1 = ld_frag_lds(bt1 + ((long)q << 13), 8);
            #pragma unroll
            for (int m = 0; m < 4; ++m) {
                acc[m][0] = __builtin_amdgcn_wmma_f32_16x16x32_bf16(
                    false, A[m], false, B0, (short)0, acc[m][0], false, false);
                acc[m][1] = __builtin_amdgcn_wmma_f32_16x16x32_bf16(
                    false, A[m], false, B1, (short)0, acc[m][1], false, false);
            }
        }
    }

    // Epilogue: scale by rdemod and store.  C/D layout: col = lane&15,
    // row = vgpr_idx + 8*(lane>=16).
    const float rd0 = rdemod[((long)b << 8) + fbase + nf];
    const float rd1 = rdemod[((long)b << 8) + fbase + 16 + nf];
    #pragma unroll
    for (int m = 0; m < 4; ++m) {
        const long rowbase =
            ((((long)b * Hd + h) * Wd + (w0 + m * 16 + (hilane << 3))) << 8)
            + fbase + nf;
        #pragma unroll
        for (int i = 0; i < 8; ++i) {
            const long o0 = rowbase + ((long)i << 8);       // row i of tile m
            out[o0]      = acc[m][0][i] * rd0;
            out[o0 + 16] = acc[m][1][i] * rd1;
        }
    }
}

// ---------------------------------------------------------------------------
extern "C" void kernel_launch(void* const* d_in, const int* in_sizes, int n_in,
                              void* d_out, int out_size, void* d_ws, size_t ws_size,
                              hipStream_t stream) {
    const float* x     = (const float*)d_in[0];   // [8,128,128,256]
    const float* style = (const float*)d_in[1];   // [8,1,1,256]
    const float* kern  = (const float*)d_in[2];   // [3,3,256,256]
    float* out = (float*)d_out;

    unsigned short* xs  = (unsigned short*)d_ws;            // 67,108,864 B
    unsigned short* wpk = xs + (size_t)Bn * Hd * Wd * Cc;   // +1,179,648 B
    float* rd = (float*)(wpk + (size_t)9 * 8 * Ff * 32);    // +8,192 B

    const int total8 = Bn * Hd * Wd * Cc / 8;               // 4,194,304
    k_prescale<<<total8 / 256, 256, 0, stream>>>(x, style, xs, total8);

    const int wtot = 9 * 8 * Ff * 32;                       // 589,824
    k_wpack<<<wtot / 256, 256, 0, stream>>>(kern, wpk, wtot);

    k_demod<<<(Bn * Ff) / 256, 256, 0, stream>>>(kern, style, rd);

    const int grid = Bn * Hd * (Wd / MT);                   // 2,048
    k_conv<<<grid, 256, 0, stream>>>(xs, wpk, rd, out);
}
